// Transformer_63376537420016
// MI455X (gfx1250) — matrix operand order repaired and checked by reading, not compiled, not run on hardware
//
#include <hip/hip_runtime.h>
#include <hip/hip_bf16.h>

// ---------------- model constants (match reference) ----------------
constexpr int NL  = 2;      // layers
constexpr int SEQ = 1024;   // tokens (B*S, B=1)
constexpr int DIM = 1024;   // model dim
constexpr int NH  = 16;     // query heads
constexpr int HDIM= 64;     // head dim
constexpr int NKV = 4;      // kv heads
constexpr int NE  = 16;     // experts
constexpr int FI  = 768;    // ffn intermediate
constexpr int VOC = 32000;  // vocab
constexpr float EPS = 1e-6f;
constexpr float ROPE_BASE = 1000000.0f;

// ---------------- WMMA / vector types ----------------
typedef __attribute__((ext_vector_type(16))) __bf16 bf16x16;
typedef __attribute__((ext_vector_type(8)))  float  f32x8;
typedef __attribute__((ext_vector_type(4)))  int    i32x4;
typedef __attribute__((ext_vector_type(8)))  int    i32x8;

union BFrag { bf16x16 v; unsigned short u[16]; unsigned w[8]; };

// Pack two f32 into two bf16 (round-half-up via +0x8000, then byte-perm).
// One v_perm_b32 + two v_add per PAIR instead of ~8 VALU ops.
__device__ __forceinline__ unsigned pkbf(float lo, float hi) {
  unsigned bl = __float_as_uint(lo) + 0x8000u;
  unsigned bh = __float_as_uint(hi) + 0x8000u;
  return __builtin_amdgcn_perm(bh, bl, 0x07060302u); // {bh[3],bh[2],bl[3],bl[2]}
}

// ============================================================================
// TDM-staged GEMM for the weight-streaming case (lak==1, lbk==1, dims %64==0).
// C[m,n] = beta*C + rowscale[m]*sum_k A[m,k]*B[n,k]
// Block = 64(m) x 64(n). TDM copies the 64x64 fp32 B-tile global->LDS
// (tensor_load_to_lds, 2-group D#), waves consume it as bf16 fragments.
// Each wave: 2 M-tiles x 1 N-tile, KC=64 chunk fully unrolled -> 4 WMMA/chunk.
// Batched over blockIdx.z (GQA via bdiv).
// ============================================================================
__global__ void __launch_bounds__(256)
gemm_bf16_wmma_tdm(const float* __restrict__ A, long la,
                   const float* __restrict__ B, long lbn,
                   float* __restrict__ C, long lc,
                   const float* __restrict__ rowscale, long rss,
                   float alpha, float beta,
                   int M, int N, int K,
                   long abatch, long bbatch, int bdiv, long cbatch)
{
  extern __shared__ float shb[];            // [64][64] B tile; dynamic LDS => lds_addr 0
  constexpr int KC = 64;
  const int wave = threadIdx.x >> 5;
  const int lane = threadIdx.x & 31;
  const int half = lane >> 4;
  const int l16  = lane & 15;
  const int z = blockIdx.z;
  A += (long)z * abatch;
  B += (long)(z / bdiv) * bbatch;
  C += (long)z * cbatch;

  const int mb = blockIdx.y * 64 + (wave >> 2) * 32;  // this wave: rows mb..mb+31
  const int nb = blockIdx.x * 64;                     // block's N window
  const int nw = (wave & 3) * 16;                     // wave's N tile in window
  const float* arow0 = A + (long)(mb + l16) * la;
  const float* arow1 = A + (long)(mb + 16 + l16) * la;
  const float* bsrc  = shb + (long)(nw + l16) * KC;   // LDS column for this lane

  f32x8 acc0 = {}, acc1 = {};

  for (int kc = 0; kc < K; kc += KC) {
    if (wave == 0) {
      // ---- build Tensor DMA Descriptor (D#), groups 0 and 1 (2-D tensor) ----
      unsigned long long ga =
          (unsigned long long)(const void*)(B + (long)nb * lbn + kc);
      i32x4 g0;
      g0[0] = 1;                                         // count=1, user mode
      g0[1] = 0;                                         // lds_addr = 0
      g0[2] = (int)(unsigned)(ga & 0xFFFFFFFFull);       // global_addr[31:0]
      g0[3] = (int)(((unsigned)(ga >> 32) & 0x01FFFFFFu) // global_addr[56:32]
                    | 0x80000000u);                      // type=2 ("image")
      i32x8 g1;
      g1[0] = (2 << 16);                  // data_size = 4B; no multicast/pad/iter
      g1[1] = (KC & 0xFFFF) << 16;        // tensor_dim0[15:0]
      g1[2] = (KC >> 16) | (64 << 16);    // tensor_dim0[31:16] | tensor_dim1[15:0]
      g1[3] = (KC << 16);                 // tensor_dim1[31:16]=0 | tile_dim0=KC
      g1[4] = 64;                         // tile_dim1 = 64 rows, tile_dim2 = 0
      g1[5] = (int)(unsigned)(lbn & 0xFFFFFFFFl);        // dim0_stride[31:0]
      g1[6] = (int)((lbn >> 32) & 0xFFFFl);              // dim0_stride[47:32]
      g1[7] = 0;
      asm volatile("tensor_load_to_lds %0, %1" :: "s"(g0), "s"(g1) : "memory");
      __builtin_amdgcn_s_wait_tensorcnt(0);
    }
    __syncthreads();                       // publish B tile to all waves

    #pragma unroll
    for (int kb = 0; kb < KC; kb += 32) {
      // A 16x32 bf16 layout: lanes<16 K {0..7,16..23}, lanes>=16 {8..15,24..31}
      const int ka1 = kc + kb + half * 8;
      const int ka2 = kc + kb + 16 + half * 8;
      // B 32x16 layout: same K order as A
      const int kb1 = kb + half * 8;

      BFrag af0, af1, bf;
      f32x8 a0 = *(const f32x8*)(arow0 + ka1);
      f32x8 a1 = *(const f32x8*)(arow0 + ka2);
      f32x8 a2 = *(const f32x8*)(arow1 + ka1);
      f32x8 a3 = *(const f32x8*)(arow1 + ka2);
      f32x8 b0 = *(const f32x8*)(bsrc + kb1);       // ds_load from TDM-staged tile
      f32x8 b1 = *(const f32x8*)(bsrc + kb1 + 16);
      #pragma unroll
      for (int j = 0; j < 4; ++j) {
        af0.w[j]     = pkbf(a0[2*j], a0[2*j+1]);
        af0.w[j + 4] = pkbf(a1[2*j], a1[2*j+1]);
        af1.w[j]     = pkbf(a2[2*j], a2[2*j+1]);
        af1.w[j + 4] = pkbf(a3[2*j], a3[2*j+1]);
        bf.w[j]      = pkbf(b0[2*j], b0[2*j+1]);
        bf.w[j + 4]  = pkbf(b1[2*j], b1[2*j+1]);
      }
      acc0 = __builtin_amdgcn_wmma_f32_16x16x32_bf16(false, af0.v, false, bf.v,
                                                     (short)0, acc0, false, false);
      acc1 = __builtin_amdgcn_wmma_f32_16x16x32_bf16(false, af1.v, false, bf.v,
                                                     (short)0, acc1, false, false);
    }
    __syncthreads();                       // all waves done before next TDM overwrite
  }

  // C layout: VGPR r -> M = r + 8*half, N = lane%16
  const int n = nb + nw + l16;
  #pragma unroll
  for (int r = 0; r < 8; ++r) {
    const int m0 = mb + half * 8 + r;
    const int m1 = m0 + 16;
    long c0 = (long)m0 * lc + n;
    long c1 = (long)m1 * lc + n;
    float v0 = alpha * acc0[r];
    float v1 = alpha * acc1[r];
    if (rowscale) { v0 *= rowscale[(long)m0 * rss]; v1 *= rowscale[(long)m1 * rss]; }
    if (beta != 0.0f) { v0 += beta * C[c0]; v1 += beta * C[c1]; }
    C[c0] = v0;
    C[c1] = v1;
  }
}

// ============================================================================
// Generic strided GEMM (fallback: gate N=16, P@V with K-strided B).
// ============================================================================
__global__ void __launch_bounds__(256)
gemm_bf16_wmma(const float* __restrict__ A, long la, long lak,
               const float* __restrict__ B, long lbn, long lbk,
               float* __restrict__ C, long lc,
               const float* __restrict__ rowscale, long rss,
               float alpha, float beta,
               int M, int N, int K,
               long abatch, long bbatch, int bdiv, long cbatch)
{
  const int wave = threadIdx.x >> 5;
  const int lane = threadIdx.x & 31;
  const int half = lane >> 4;
  const int l16  = lane & 15;
  const int m0 = (blockIdx.y * 2 + (wave >> 2)) * 16;
  const int n0 = (blockIdx.x * 4 + (wave & 3)) * 16;
  if (n0 >= N || m0 >= M) return;           // wave-uniform: EXEC stays all-ones

  const int z = blockIdx.z;
  A += (long)z * abatch;
  B += (long)(z / bdiv) * bbatch;
  C += (long)z * cbatch;

  const float* arow = A + (long)(m0 + l16) * la;
  const float* bcol = B + (long)(n0 + l16) * lbn;

  f32x8 acc = {};
  for (int kb = 0; kb < K; kb += 32) {
    BFrag af, bf;
    const int ka1 = kb + half * 8;
    const int ka2 = kb + 16 + half * 8;
    const int kb1 = kb + half * 8;

    if (lak == 1) {
      f32x8 a0 = *(const f32x8*)(arow + ka1);
      f32x8 a1 = *(const f32x8*)(arow + ka2);
      #pragma unroll
      for (int j = 0; j < 4; ++j) {
        af.w[j]     = pkbf(a0[2*j], a0[2*j+1]);
        af.w[j + 4] = pkbf(a1[2*j], a1[2*j+1]);
      }
    } else {
      #pragma unroll
      for (int j = 0; j < 4; ++j) {
        af.w[j]     = pkbf(arow[(long)(ka1 + 2*j) * lak], arow[(long)(ka1 + 2*j + 1) * lak]);
        af.w[j + 4] = pkbf(arow[(long)(ka2 + 2*j) * lak], arow[(long)(ka2 + 2*j + 1) * lak]);
      }
    }
    if (lbk == 1) {
      __builtin_prefetch(bcol + kb1 + 256, 0, 1);
      f32x8 b0 = *(const f32x8*)(bcol + kb1);
      f32x8 b1 = *(const f32x8*)(bcol + kb1 + 16);
      #pragma unroll
      for (int j = 0; j < 4; ++j) {
        bf.w[j]     = pkbf(b0[2*j], b0[2*j+1]);
        bf.w[j + 4] = pkbf(b1[2*j], b1[2*j+1]);
      }
    } else {
      #pragma unroll
      for (int j = 0; j < 8; ++j)
        bf.w[j] = pkbf(bcol[(long)(kb1 + 2*j + (j >= 4 ? 8 : 0)) * lbk], bcol[(long)(kb1 + 2*j + 1 + (j >= 4 ? 8 : 0)) * lbk]);
    }

    acc = __builtin_amdgcn_wmma_f32_16x16x32_bf16(false, af.v, false, bf.v,
                                                  (short)0, acc, false, false);
  }

  #pragma unroll
  for (int r = 0; r < 8; ++r) {
    const int m = m0 + half * 8 + r;
    const long ci = (long)m * lc + n0 + l16;
    float v = alpha * acc[r];
    if (rowscale) v *= rowscale[(long)m * rss];
    if (beta != 0.0f) v += beta * C[ci];
    C[ci] = v;
  }
}

// ============================================================================
// Embedding gather: x[t,:] = tok_emb[idx[t],:]
// ============================================================================
__global__ void embed_kernel(const int* __restrict__ idx, const float* __restrict__ emb,
                             float* __restrict__ x, int C)
{
  const long t = blockIdx.x;
  const float* src = emb + (long)idx[t] * C;
  float* dst = x + t * (long)C;
  for (int c = threadIdx.x; c < C; c += blockDim.x) dst[c] = src[c];
}

// ============================================================================
// Row RMSNorm: o[r,:] = x[r,:] * rsqrt(mean(x^2)+eps) * w
// ============================================================================
__global__ void rmsnorm_kernel(const float* __restrict__ x, const float* __restrict__ w,
                               float* __restrict__ o, int C)
{
  const long row = blockIdx.x;
  const float* xr = x + row * (long)C;
  float* orow = o + row * (long)C;
  __shared__ float sh[8];
  __shared__ float bc;
  float s = 0.f;
  for (int c = threadIdx.x; c < C; c += blockDim.x) { float v = xr[c]; s += v * v; }
  for (int off = 16; off; off >>= 1) s += __shfl_xor(s, off);
  if ((threadIdx.x & 31) == 0) sh[threadIdx.x >> 5] = s;
  __syncthreads();
  if (threadIdx.x < 8) {
    float v = sh[threadIdx.x];
    for (int off = 4; off; off >>= 1) v += __shfl_xor(v, off);
    if (threadIdx.x == 0) bc = rsqrtf(v / (float)C + EPS);
  }
  __syncthreads();
  const float r = bc;
  for (int c = threadIdx.x; c < C; c += blockDim.x) orow[c] = xr[c] * r * w[c];
}

// ============================================================================
// Fused per-head RMSNorm (over HDIM=64) + RoPE, in place. 64 threads/block.
// ============================================================================
__global__ void headnorm_rope(float* __restrict__ q, const float* __restrict__ nw, int nheads)
{
  const int t  = blockIdx.x / nheads;       // token position
  const int hh = blockIdx.x % nheads;
  float* r = q + ((long)t * nheads + hh) * HDIM;
  const int d = threadIdx.x;
  float v = r[d];
  float s = v * v;
  for (int off = 16; off; off >>= 1) s += __shfl_xor(s, off);
  __shared__ float sh[2];
  __shared__ float bc;
  if ((d & 31) == 0) sh[d >> 5] = s;
  __syncthreads();
  if (d == 0) bc = rsqrtf((sh[0] + sh[1]) / (float)HDIM + EPS);
  __syncthreads();
  float n = v * bc * nw[d];
  const int j = d >> 1;
  const float inv = __powf(ROPE_BASE, -(float)(2 * j) / (float)HDIM);
  float sn, cs;
  __sincosf((float)t * inv, &sn, &cs);
  const float other = __shfl_xor(n, 1);
  r[d] = (d & 1) ? (n * cs + other * sn) : (n * cs - other * sn);
}

// ============================================================================
// Causal softmax over row [h*S + s1] of scores, scale folded in, in place.
// ============================================================================
__global__ void softmax_causal(float* __restrict__ p, int Slen, float scale)
{
  const long row = blockIdx.x;
  const int s1 = (int)(row % Slen);
  float* r = p + row * (long)Slen;
  const int tid = threadIdx.x;
  __shared__ float sh[8];
  __shared__ float bc;
  float m = -1e30f;
  for (int c = tid; c <= s1; c += blockDim.x) m = fmaxf(m, r[c] * scale);
  for (int off = 16; off; off >>= 1) m = fmaxf(m, __shfl_xor(m, off));
  if ((tid & 31) == 0) sh[tid >> 5] = m;
  __syncthreads();
  if (tid < 8) {
    float v = sh[tid];
    for (int off = 4; off; off >>= 1) v = fmaxf(v, __shfl_xor(v, off));
    if (tid == 0) bc = v;
  }
  __syncthreads();
  m = bc;
  __syncthreads();
  float s = 0.f;
  for (int c = tid; c <= s1; c += blockDim.x) { float e = __expf(r[c] * scale - m); r[c] = e; s += e; }
  for (int off = 16; off; off >>= 1) s += __shfl_xor(s, off);
  if ((tid & 31) == 0) sh[tid >> 5] = s;
  __syncthreads();
  if (tid < 8) {
    float v = sh[tid];
    for (int off = 4; off; off >>= 1) v += __shfl_xor(v, off);
    if (tid == 0) bc = v;
  }
  __syncthreads();
  const float invs = 1.0f / bc;
  for (int c = tid; c < Slen; c += blockDim.x) r[c] = (c <= s1) ? r[c] * invs : 0.0f;
}

// ============================================================================
// h = silu(a) * b  (grid-stride)
// ============================================================================
__global__ void silu_mul(const float* __restrict__ a, const float* __restrict__ b,
                         float* __restrict__ o, long n)
{
  long i = (long)blockIdx.x * blockDim.x + threadIdx.x;
  const long stride = (long)gridDim.x * blockDim.x;
  for (; i < n; i += stride) { float x = a[i]; o[i] = (x / (1.0f + __expf(-x))) * b[i]; }
}

// ============================================================================
// Gating: softmax over E=16 logits, top-8, renormalize -> coef[t,e] (0 if unpicked)
// ============================================================================
__global__ void gating_kernel(const float* __restrict__ glog, float* __restrict__ coef, int T)
{
  const int t = blockIdx.x * blockDim.x + threadIdx.x;
  if (t >= T) return;
  const float* g = glog + (long)t * NE;
  float p[NE];
  float m = -1e30f;
  for (int e = 0; e < NE; ++e) m = fmaxf(m, g[e]);
  float s = 0.f;
  for (int e = 0; e < NE; ++e) { p[e] = __expf(g[e] - m); s += p[e]; }
  for (int e = 0; e < NE; ++e) p[e] /= s;
  bool sel[NE];
  for (int e = 0; e < NE; ++e) sel[e] = false;
  float ts = 0.f;
  for (int k = 0; k < 8; ++k) {
    int bi = 0; float bv = -1.0f;
    for (int e = 0; e < NE; ++e) if (!sel[e] && p[e] > bv) { bv = p[e]; bi = e; }
    sel[bi] = true; ts += bv;
  }
  float* c = coef + (long)t * NE;
  for (int e = 0; e < NE; ++e) c[e] = sel[e] ? p[e] / ts : 0.0f;
}

// ---------------- host-side GEMM dispatcher ----------------
static inline void gemm(hipStream_t st, const float* A, long la, long lak,
                        const float* B, long lbn, long lbk,
                        float* C, long lc, const float* rs, long rss,
                        float alpha, float beta, int M, int N, int K,
                        long ab, long bb, int bdiv, long cb, int Z)
{
  if (lak == 1 && lbk == 1 && (M % 64) == 0 && (N % 64) == 0 && (K % 64) == 0) {
    dim3 grid((unsigned)(N / 64), (unsigned)(M / 64), (unsigned)Z);
    gemm_bf16_wmma_tdm<<<grid, 256, 64 * 64 * sizeof(float), st>>>(
        A, la, B, lbn, C, lc, rs, rss, alpha, beta, M, N, K, ab, bb, bdiv, cb);
  } else {
    dim3 grid((unsigned)((N + 63) / 64), (unsigned)(M / 32), (unsigned)Z);
    gemm_bf16_wmma<<<grid, 256, 0, st>>>(A, la, lak, B, lbn, lbk, C, lc, rs, rss,
                                         alpha, beta, M, N, K, ab, bb, bdiv, cb);
  }
}

extern "C" void kernel_launch(void* const* d_in, const int* in_sizes, int n_in,
                              void* d_out, int out_size, void* d_ws, size_t ws_size,
                              hipStream_t stream)
{
  const int*   idx          = (const int*)  d_in[0];
  const float* tok_emb      = (const float*)d_in[1];
  const float* q_w          = (const float*)d_in[2];
  const float* k_w          = (const float*)d_in[3];
  const float* v_w          = (const float*)d_in[4];
  const float* o_w          = (const float*)d_in[5];
  const float* qn_w         = (const float*)d_in[6];
  const float* kn_w         = (const float*)d_in[7];
  const float* attn_norm_w  = (const float*)d_in[8];
  const float* ffn_norm_w   = (const float*)d_in[9];
  const float* gate_w       = (const float*)d_in[10];
  const float* w1           = (const float*)d_in[11];
  const float* w2           = (const float*)d_in[12];
  const float* w3           = (const float*)d_in[13];
  const float* final_norm_w = (const float*)d_in[14];
  const float* out_w        = (const float*)d_in[15];
  float* out = (float*)d_out;

  // ---- workspace carve-up (all fp32; WMMA converts to bf16 at load) ----
  float* ws = (float*)d_ws;
  float* x      = ws; ws += (long)SEQ * DIM;          // residual stream
  float* h      = ws; ws += (long)SEQ * DIM;          // rmsnorm output
  float* qf     = ws; ws += (long)SEQ * NH  * HDIM;   // q (post proj / rope, in place)
  float* kf     = ws; ws += (long)SEQ * NKV * HDIM;
  float* vf     = ws; ws += (long)SEQ * NKV * HDIM;
  float* scores = ws; ws += (long)NH * SEQ * SEQ;     // attn scores -> probs in place
  float* y      = ws; ws += (long)SEQ * NH * HDIM;    // attn output
  float* glog   = ws; ws += (long)SEQ * NE;
  float* coef   = ws; ws += (long)SEQ * NE;
  float* x1     = ws; ws += (long)SEQ * FI;
  float* x3     = ws; ws += (long)SEQ * FI;
  const size_t need = (size_t)(ws - (float*)d_ws) * sizeof(float);
  if (ws_size < need) return;

  // ---- embedding ----
  embed_kernel<<<SEQ, 256, 0, stream>>>(idx, tok_emb, x, DIM);

  for (int l = 0; l < NL; ++l) {
    // attention pre-norm
    rmsnorm_kernel<<<SEQ, 256, 0, stream>>>(x, attn_norm_w + (long)l * DIM, h, DIM);

    // QKV projections (C = h @ W^T)  [TDM path]
    gemm(stream, h, DIM, 1, q_w + (long)l * NH  * HDIM * DIM, DIM, 1,
         qf, NH * HDIM, nullptr, 0, 1.f, 0.f, SEQ, NH * HDIM, DIM, 0, 0, 1, 0, 1);
    gemm(stream, h, DIM, 1, k_w + (long)l * NKV * HDIM * DIM, DIM, 1,
         kf, NKV * HDIM, nullptr, 0, 1.f, 0.f, SEQ, NKV * HDIM, DIM, 0, 0, 1, 0, 1);
    gemm(stream, h, DIM, 1, v_w + (long)l * NKV * HDIM * DIM, DIM, 1,
         vf, NKV * HDIM, nullptr, 0, 1.f, 0.f, SEQ, NKV * HDIM, DIM, 0, 0, 1, 0, 1);

    // per-head rmsnorm + rope
    headnorm_rope<<<SEQ * NH,  64, 0, stream>>>(qf, qn_w + (long)l * HDIM, NH);
    headnorm_rope<<<SEQ * NKV, 64, 0, stream>>>(kf, kn_w + (long)l * HDIM, NKV);

    // scores[z] = q_z @ k_{z/4}^T  (batched over heads, GQA bdiv=4) [TDM path]
    gemm(stream, qf, NH * HDIM, 1, kf, NKV * HDIM, 1,
         scores, SEQ, nullptr, 0, 1.f, 0.f,
         SEQ, SEQ, HDIM, HDIM, HDIM, NH / NKV, (long)SEQ * SEQ, NH);

    // scale + causal mask + softmax (in place)
    softmax_causal<<<NH * SEQ, 256, 0, stream>>>(scores, SEQ, 0.125f);

    // y[z] = P_z @ v_{z/4}  (B accessed K-strided -> generic path)
    gemm(stream, scores, SEQ, 1, vf, 1, NKV * HDIM,
         y, NH * HDIM, nullptr, 0, 1.f, 0.f,
         SEQ, HDIM, SEQ, (long)SEQ * SEQ, HDIM, NH / NKV, HDIM, NH);

    // x += y @ o_w^T  [TDM path]
    gemm(stream, y, NH * HDIM, 1, o_w + (long)l * DIM * NH * HDIM, NH * HDIM, 1,
         x, DIM, nullptr, 0, 1.f, 1.f, SEQ, DIM, NH * HDIM, 0, 0, 1, 0, 1);

    // ---- MoE ----
    rmsnorm_kernel<<<SEQ, 256, 0, stream>>>(x, ffn_norm_w + (long)l * DIM, h, DIM);
    gemm(stream, h, DIM, 1, gate_w + (long)l * NE * DIM, DIM, 1,
         glog, NE, nullptr, 0, 1.f, 0.f, SEQ, NE, DIM, 0, 0, 1, 0, 1);  // generic (N=16)
    gating_kernel<<<SEQ / 256, 256, 0, stream>>>(glog, coef, SEQ);

    for (int e = 0; e < NE; ++e) {
      const float* w1e = w1 + ((long)l * NE + e) * (long)FI * DIM;
      const float* w3e = w3 + ((long)l * NE + e) * (long)FI * DIM;
      const float* w2e = w2 + ((long)l * NE + e) * (long)DIM * FI;
      gemm(stream, h, DIM, 1, w1e, DIM, 1, x1, FI, nullptr, 0, 1.f, 0.f,
           SEQ, FI, DIM, 0, 0, 1, 0, 1);
      gemm(stream, h, DIM, 1, w3e, DIM, 1, x3, FI, nullptr, 0, 1.f, 0.f,
           SEQ, FI, DIM, 0, 0, 1, 0, 1);
      silu_mul<<<1024, 256, 0, stream>>>(x1, x3, x1, (long)SEQ * FI);
      // x += coef[:,e] * (silu(x1)*x3) @ w2_e^T  (rowscale epilogue) [TDM path]
      gemm(stream, x1, FI, 1, w2e, FI, 1, x, DIM, coef + e, NE, 1.f, 1.f,
           SEQ, DIM, FI, 0, 0, 1, 0, 1);
    }
  }

  // final norm + vocab projection [TDM path]
  rmsnorm_kernel<<<SEQ, 256, 0, stream>>>(x, final_norm_w, h, DIM);
  gemm(stream, h, DIM, 1, out_w, DIM, 1, out, VOC, nullptr, 0, 1.f, 0.f,
       SEQ, VOC, DIM, 0, 0, 1, 0, 1);
}
